// GNN_12060268167169
// MI455X (gfx1250) — matrix operand order repaired and checked by reading, not compiled, not run on hardware
//
#include <hip/hip_runtime.h>
#include <math.h>

typedef __attribute__((ext_vector_type(16))) _Float16 v16h;
typedef __attribute__((ext_vector_type(8)))  float    v8f;

#define NEGKEY 0x007FFFFFu   // key(-inf)

__device__ __forceinline__ v8f v8f_zero() {
  v8f z;
#pragma unroll
  for (int i = 0; i < 8; ++i) z[i] = 0.f;
  return z;
}

__device__ __forceinline__ v8f wmma_f16(v16h a, v16h b, v8f c) {
  return __builtin_amdgcn_wmma_f32_16x16x32_f16(false, a, false, b, (short)0, c, false, false);
}

// Fragment-linear layouts: a 16x32 (A) or 32x16 (B) tile is 512 halves laid out
// as [lane][h] so each lane's 16 halves are contiguous (32B -> 2x b128 loads).
__device__ __forceinline__ v16h load_frag_lds(const _Float16* swz, int tile, int lane) {
  return *(const v16h*)(swz + ((tile * 32 + lane) << 4));
}
__device__ __forceinline__ v16h load_frag_g(const _Float16* w, int tile, int lane) {
  return *(const v16h*)(w + (((size_t)tile * 32 + lane) << 4));
}

// ISA 7.12.2 A-frag mapping: element (m, k) -> lane = m + 16*((k>>3)&1),
// h = ((k>>4)<<3) + (k&7).  4 consecutive k (k%4==0) => 4 consecutive h, one lane.
__device__ __forceinline__ int a_swz_idx(int m, int c4) {
  int kt = c4 >> 5, kk = c4 & 31;
  int lane_t = m + (((kk >> 3) & 1) << 4);
  int h0 = ((kk >> 4) << 3) + (kk & 7);
  return ((kt * 32 + lane_t) << 4) + h0;
}

// CDNA5 async DMA: global -> LDS, 8 bytes per lane, tracked by ASYNCcnt.
__device__ __forceinline__ void async_copy_b64(_Float16* lds_dst, const _Float16* gsrc) {
  unsigned ldsoff = (unsigned)(uintptr_t)lds_dst;
  asm volatile("global_load_async_to_lds_b64 %0, %1, off"
               :: "v"(ldsoff), "v"(gsrc) : "memory");
}
__device__ __forceinline__ void wait_async0() {
  asm volatile("s_wait_asynccnt 0" ::: "memory");
}

__device__ __forceinline__ unsigned fkey(float v) {
  unsigned b = __float_as_uint(v);
  return (b & 0x80000000u) ? ~b : (b | 0x80000000u);
}
__device__ __forceinline__ float funkey(unsigned k) {
  if (k == NEGKEY) return 0.f;  // no incoming edge -> 0
  unsigned b = (k & 0x80000000u) ? (k & 0x7FFFFFFFu) : ~k;
  return __uint_as_float(b);
}
// fast transcendentals: v_exp_f32 + v_rcp_f32 instead of precise IEEE division
__device__ __forceinline__ float fast_sigmoid(float x) {
  return __builtin_amdgcn_rcpf(1.f + __expf(-x));
}
__device__ __forceinline__ float fast_tanh(float x) {
  return 1.f - 2.f * __builtin_amdgcn_rcpf(__expf(2.f * x) + 1.f);
}
__device__ __forceinline__ float gelu_tanh(float x) {
  return 0.5f * x * (1.f + fast_tanh(0.7978845608028654f * (x + 0.044715f * x * x * x)));
}

// ------- weight convert: f32 -> f16, padded, emitted in FRAGMENT-LINEAR layout
__global__ void k_convfrag(_Float16* dst, const float* src, int K, int Nc,
                           int Kp, int Np, int tr) {
  int i = blockIdx.x * blockDim.x + threadIdx.x;
  if (i >= Kp * Np) return;
  int h = i & 15, lane = (i >> 4) & 31, tile = i >> 9;
  int NT = Np >> 4;
  int nt = tile % NT, kt = tile / NT;
  int n = nt * 16 + (lane & 15);
  int k = kt * 32 + ((lane >> 4) << 3) + ((h >> 3) << 4) + (h & 7);   // B-frag: table order
  float v = 0.f;
  if (k < K && n < Nc) v = tr ? src[n * K + k] : src[k * Nc + n];
  dst[i] = (_Float16)v;
}

__global__ void k_convh(_Float16* dst, const float* src, int n) {
  for (int i = blockIdx.x * blockDim.x + threadIdx.x; i < n; i += gridDim.x * blockDim.x)
    dst[i] = (_Float16)src[i];
}

__global__ void k_fillu(unsigned* p, int n, unsigned v) {
  for (int i = blockIdx.x * blockDim.x + threadIdx.x; i < n; i += gridDim.x * blockDim.x)
    p[i] = v;
}

// ---------------- GRU: 128 thr = 4 waves, wave-private 16 nodes, no barriers.
// Gates are processed one 16-column feature tile at a time so only 6 WMMA
// accumulators are live at once (low VGPR pressure -> pipelined B-frag loads).
__global__ __launch_bounds__(128) void k_gru(
    const int* __restrict__ tokens, const _Float16* __restrict__ embedh,
    const _Float16* __restrict__ WiF, const _Float16* __restrict__ WhF,  // frag-linear [2][12][512]
    const float* __restrict__ bi, const float* __restrict__ bh,
    float* __restrict__ hout, int N) {
  __shared__ __align__(32) _Float16 xswz[4][2 * 512];
  __shared__ __align__(32) _Float16 hswz[4][2 * 512];
  const int tid = threadIdx.x, lane = tid & 31, wave = tid >> 5;
  const int nbase = blockIdx.x * 64 + wave * 16;
  _Float16* xs = xswz[wave];
  _Float16* hs = hswz[wave];
  const int n = lane & 15, mofs = (lane >> 4) << 3;

  for (int i = lane; i < 2 * 512 / 4; i += 32) ((unsigned long long*)hs)[i] = 0ull;

  v8f hreg[4];
#pragma unroll
  for (int j = 0; j < 4; ++j) hreg[j] = v8f_zero();

  float biR[4], biZ[4], biN[4], bhR[4], bhZ[4], bhN[4];
#pragma unroll
  for (int ft = 0; ft < 4; ++ft) {
    int c = ft * 16 + n;
    biR[ft] = bi[c]; biZ[ft] = bi[64 + c]; biN[ft] = bi[128 + c];
    bhR[ft] = bh[c]; bhZ[ft] = bh[64 + c]; bhN[ft] = bh[128 + c];
  }

  for (int s = 0; s < 8; ++s) {
    // async-gather this step's x (16 rows x 64 halves) into swizzled LDS.
    // In-order issue + the s_wait_dscnt before the previous step's WMMA
    // guarantees the old xs contents were consumed before these land.
    for (int i = lane; i < 16 * 16; i += 32) {
      int row = i >> 4, c4 = (i & 15) << 2;
      int node = nbase + row;
      if (node < N) {
        int tok = tokens[node * 8 + s];
        async_copy_b64(xs + a_swz_idx(row, c4), embedh + tok * 64 + c4);
      }
    }
    wait_async0();

    v16h ax0 = load_frag_lds(xs, 0, lane), ax1 = load_frag_lds(xs, 1, lane);
    v16h ah0 = load_frag_lds(hs, 0, lane), ah1 = load_frag_lds(hs, 1, lane);

#pragma unroll
    for (int ft = 0; ft < 4; ++ft) {
      v8f gr = v8f_zero(), gz = v8f_zero(), gn = v8f_zero();
      v8f hr = v8f_zero(), hz = v8f_zero(), hn = v8f_zero();
#pragma unroll
      for (int kt = 0; kt < 2; ++kt) {
        v16h ax = kt ? ax1 : ax0;
        v16h ah = kt ? ah1 : ah0;
        gr = wmma_f16(ax, load_frag_g(WiF, kt * 12 + ft,     lane), gr);
        gz = wmma_f16(ax, load_frag_g(WiF, kt * 12 + 4 + ft, lane), gz);
        gn = wmma_f16(ax, load_frag_g(WiF, kt * 12 + 8 + ft, lane), gn);
        hr = wmma_f16(ah, load_frag_g(WhF, kt * 12 + ft,     lane), hr);
        hz = wmma_f16(ah, load_frag_g(WhF, kt * 12 + 4 + ft, lane), hz);
        hn = wmma_f16(ah, load_frag_g(WhF, kt * 12 + 8 + ft, lane), hn);
      }
      // gates in C-frag layout: (lane, r) -> m = r + 8*(lane>=16), col = ft*16 + n
      int c = ft * 16 + n;
#pragma unroll
      for (int r = 0; r < 8; ++r) {
        float irv = gr[r] + biR[ft], hrv = hr[r] + bhR[ft];
        float izv = gz[r] + biZ[ft], hzv = hz[r] + bhZ[ft];
        float inv = gn[r] + biN[ft], hnv = hn[r] + bhN[ft];
        float rr = fast_sigmoid(irv + hrv);
        float zz = fast_sigmoid(izv + hzv);
        float nn = fast_tanh(inv + rr * hnv);
        float hv = (1.f - zz) * nn + zz * hreg[ft][r];
        hreg[ft][r] = hv;
        int m = r + mofs, kt = c >> 5, kk = c & 31;
        hs[((kt * 32 + (m + (((kk >> 3) & 1) << 4))) << 4) +
           ((kk >> 4) << 3) + (kk & 7)] = (_Float16)hv;
      }
    }
  }
#pragma unroll
  for (int ft = 0; ft < 4; ++ft)
#pragma unroll
    for (int r = 0; r < 8; ++r) {
      int node = nbase + r + mofs;
      if (node < N) hout[node * 64 + ft * 16 + n] = hreg[ft][r];
    }
}

// -------- build node features: onehot(type,6) ++ select(GRU) ++ zero pad ----
__global__ void k_combine(const float* __restrict__ hmap, const float* __restrict__ hmem,
                          const int* __restrict__ type, _Float16* __restrict__ dst,
                          int N, int stride) {
  for (int i = blockIdx.x * blockDim.x + threadIdx.x; i < N * stride;
       i += gridDim.x * blockDim.x) {
    int nd = i / stride, c = i % stride;
    int t = type[nd];
    float v = 0.f;
    if (c < 6) v = (t == c) ? 1.f : 0.f;
    else if (c < 70) {
      int d = c - 6;
      v = (t == 0) ? hmap[nd * 64 + d] : ((t == 5) ? hmem[nd * 64 + d] : 0.f);
    }
    dst[i] = (_Float16)v;
  }
}

__global__ void k_build_u(const _Float16* __restrict__ h, int hs, int nA,
                          const unsigned* __restrict__ keys, int aggW, int nB,
                          _Float16* __restrict__ dst, int ds, int N) {
  for (int i = blockIdx.x * blockDim.x + threadIdx.x; i < N * ds;
       i += gridDim.x * blockDim.x) {
    int nd = i / ds, c = i % ds;
    float v = 0.f;
    if (c < nA) v = (float)h[nd * hs + c];
    else if (c < nA + nB) v = funkey(keys[(size_t)nd * aggW + (c - nA)]);
    dst[i] = (_Float16)v;
  }
}

__global__ void k_concat2(const _Float16* __restrict__ A, int As, int nA,
                          const _Float16* __restrict__ B, int Bs, int nB,
                          _Float16* __restrict__ dst, int ds, int N) {
  for (int i = blockIdx.x * blockDim.x + threadIdx.x; i < N * ds;
       i += gridDim.x * blockDim.x) {
    int nd = i / ds, c = i % ds;
    _Float16 v = (_Float16)0.f;
    if (c < nA) v = A[nd * As + c];
    else if (c < nA + nB) v = B[nd * Bs + (c - nA)];
    dst[i] = v;
  }
}

// -------- per-edge message GEMM + scatter-max ------------------------------
template <int KP, int NP>
__global__ __launch_bounds__(128) void k_msg(
    const _Float16* __restrict__ nodebuf, int nstride,
    const _Float16* __restrict__ wm,          // [T][KP*NP] frag-linear
    const float* __restrict__ bm, int bstride,
    const int* __restrict__ src, const int* __restrict__ dstv, int E,
    int Nvalid, unsigned* __restrict__ aggK, int aggW, int act) {
  constexpr int KT = KP / 32, NT = NP / 16;
  __shared__ __align__(32) _Float16 Aswz[KT * 512];
  __shared__ int dstid[16];
  const int t = blockIdx.y, e0 = blockIdx.x * 16;
  const int tid = threadIdx.x, lane = tid & 31, wave = tid >> 5;

  if (tid < 16) {
    int e = e0 + tid;
    dstid[tid] = (e < E) ? dstv[t * E + e] : -1;
  }
  // async DMA gather of 16 edge rows into swizzled LDS (garbage rows are
  // discarded via dstid < 0; GEMM rows are independent).
  for (int i = tid; i < 16 * (KP / 4); i += 128) {
    int row = i / (KP / 4), c4 = (i % (KP / 4)) * 4;
    int e = e0 + row;
    if (e < E) {
      int sidx = src[t * E + e];
      async_copy_b64(Aswz + a_swz_idx(row, c4),
                     nodebuf + (size_t)sidx * nstride + c4);
    }
  }
  wait_async0();
  __syncthreads();

  const _Float16* wf = wm + (size_t)t * KP * NP;
  v16h afr[KT];
#pragma unroll
  for (int kt = 0; kt < KT; ++kt) afr[kt] = load_frag_lds(Aswz, kt, lane);

  for (int nt = wave; nt < NT; nt += 4) {      // wave-uniform: EXEC stays full
    v8f acc = v8f_zero();
#pragma unroll
    for (int kt = 0; kt < KT; ++kt)
      acc = wmma_f16(afr[kt], load_frag_g(wf, kt * NT + nt, lane), acc);
    const int n = lane & 15, mofs = (lane >> 4) << 3;
    const int col = nt * 16 + n;
    if (col < Nvalid) {
      float bb = bm[t * bstride + col];
#pragma unroll
      for (int r = 0; r < 8; ++r) {
        int d = dstid[r + mofs];
        if (d >= 0) {
          float v = acc[r] + bb;
          if (act) v = gelu_tanh(v);
          atomicMax(&aggK[(size_t)d * aggW + col], fkey(v));
        }
      }
    }
  }
}

// -------- dense node update GEMM -------------------------------------------
template <int KP, int NP>
__global__ __launch_bounds__(128) void k_upd(
    const _Float16* __restrict__ inbuf, int istride,
    const _Float16* __restrict__ wu, const float* __restrict__ bu,
    int Nvalid, _Float16* __restrict__ outbuf, int ostride, int N, int act) {
  constexpr int KT = KP / 32, NT = NP / 16;
  __shared__ __align__(32) _Float16 Aswz[KT * 512];
  const int tid = threadIdx.x, lane = tid & 31, wave = tid >> 5;
  const int nb = blockIdx.x * 16;

  for (int i = tid; i < 16 * (KP / 4); i += 128) {
    int row = i / (KP / 4), c4 = (i % (KP / 4)) * 4;
    int node = nb + row;
    if (node < N)
      async_copy_b64(Aswz + a_swz_idx(row, c4),
                     inbuf + (size_t)node * istride + c4);
  }
  wait_async0();
  __syncthreads();

  v16h afr[KT];
#pragma unroll
  for (int kt = 0; kt < KT; ++kt) afr[kt] = load_frag_lds(Aswz, kt, lane);

  for (int nt = wave; nt < NT; nt += 4) {
    v8f acc = v8f_zero();
#pragma unroll
    for (int kt = 0; kt < KT; ++kt)
      acc = wmma_f16(afr[kt], load_frag_g(wu, kt * NT + nt, lane), acc);
    const int n = lane & 15, mofs = (lane >> 4) << 3;
    const int col = nt * 16 + n;
    if (col < Nvalid) {
      float bb = bu[col];
#pragma unroll
      for (int r = 0; r < 8; ++r) {
        int node = nb + r + mofs;
        if (node < N) {
          float v = acc[r] + bb;
          if (act) v = fast_tanh(v);
          outbuf[(size_t)node * ostride + col] = (_Float16)v;
        }
      }
    }
  }
  for (int i = tid; i < 16 * (ostride - Nvalid); i += 128) {
    int row = i / (ostride - Nvalid), c = Nvalid + i % (ostride - Nvalid);
    int node = nb + row;
    if (node < N) outbuf[(size_t)node * ostride + c] = (_Float16)0.f;
  }
}

// -------- final 2-layer head on one node -----------------------------------
__global__ void k_head(const _Float16* __restrict__ hbuf, int hstride,
                       const int* __restrict__ midx,
                       const float* __restrict__ W1, const float* __restrict__ b1,
                       const float* __restrict__ W2, const float* __restrict__ b2,
                       float* __restrict__ out) {
  __shared__ float x[70];
  __shared__ float y[70];
  const int tid = threadIdx.x;
  const int nd = *midx;
  if (tid < 70) x[tid] = (float)hbuf[(size_t)nd * hstride + tid];
  __syncthreads();
  if (tid < 70) {
    float s = b1[tid];
    for (int d = 0; d < 70; ++d) s += x[d] * W1[d * 70 + tid];
    y[tid] = fmaxf(s, 0.f);
  }
  __syncthreads();
  for (int j = tid; j < 640; j += blockDim.x) {
    float s = b2[j];
    for (int d = 0; d < 70; ++d) s += y[d] * W2[d * 640 + j];
    out[j] = s;
  }
}

// ===========================================================================
extern "C" void kernel_launch(void* const* d_in, const int* in_sizes, int n_in,
                              void* d_out, int out_size, void* d_ws, size_t ws_size,
                              hipStream_t stream) {
  (void)in_sizes; (void)n_in; (void)out_size; (void)ws_size;
  const int N = 100000, V = 10000, H = 70, T = 2, E = 250000;
  const int NSTR = 160, USTR = 288, AGGW_P = 80, AGGW_A = 144;

  const int*   tokens   = (const int*)d_in[0];
  const int*   type_idx = (const int*)d_in[1];
  const int*   edge_src = (const int*)d_in[2];
  const int*   edge_dst = (const int*)d_in[3];
  const int*   map_idx  = (const int*)d_in[4];
  const float* embed    = (const float*)d_in[5];
  const float* Wi_map   = (const float*)d_in[6];
  const float* Wh_map   = (const float*)d_in[7];
  const float* bi_map   = (const float*)d_in[8];
  const float* bh_map   = (const float*)d_in[9];
  const float* Wi_mem   = (const float*)d_in[10];
  const float* Wh_mem   = (const float*)d_in[11];
  const float* bi_mem   = (const float*)d_in[12];
  const float* bh_mem   = (const float*)d_in[13];
  const float* Wm_plain = (const float*)d_in[14];
  const float* bm_plain = (const float*)d_in[15];
  const float* Wu_plain = (const float*)d_in[16];
  const float* bu_plain = (const float*)d_in[17];
  const float* Wm_ar    = (const float*)d_in[18];
  const float* bm_ar    = (const float*)d_in[19];
  const float* Wu_ar    = (const float*)d_in[20];
  const float* bu_ar    = (const float*)d_in[21];
  const float* W1 = (const float*)d_in[22];
  const float* b1 = (const float*)d_in[23];
  const float* W2 = (const float*)d_in[24];
  const float* b2 = (const float*)d_in[25];
  float* out = (float*)d_out;

  char* ws = (char*)d_ws; size_t off = 0;
  auto carve = [&](size_t bytes) -> char* {
    char* p = ws + off; off += (bytes + 255) & ~(size_t)255; return p;
  };
  _Float16* WiTmap = (_Float16*)carve(64 * 192 * 2);
  _Float16* WhTmap = (_Float16*)carve(64 * 192 * 2);
  _Float16* WiTmem = (_Float16*)carve(64 * 192 * 2);
  _Float16* WhTmem = (_Float16*)carve(64 * 192 * 2);
  _Float16* wmP = (_Float16*)carve((size_t)12 * 96 * 80 * 2);
  _Float16* wuP = (_Float16*)carve((size_t)6 * 160 * 80 * 2);
  _Float16* wmA = (_Float16*)carve((size_t)4 * 160 * 144 * 2);
  _Float16* wuA = (_Float16*)carve((size_t)2 * 288 * 80 * 2);
  _Float16* embedh = (_Float16*)carve((size_t)V * 64 * 2);
  float*    hmap = (float*)carve((size_t)N * 64 * 4);
  float*    hmem = (float*)carve((size_t)N * 64 * 4);
  _Float16* bufH = (_Float16*)carve((size_t)N * NSTR * 2);
  _Float16* bufS = (_Float16*)carve((size_t)N * NSTR * 2);
  _Float16* bufT = (_Float16*)carve((size_t)N * NSTR * 2);
  _Float16* ubuf = (_Float16*)carve((size_t)N * USTR * 2);
  unsigned* aggK = (unsigned*)carve((size_t)N * AGGW_A * 4);

  auto conv = [&](_Float16* dst, const float* src, int K, int Nc, int Kp, int Np, int tr) {
    k_convfrag<<<(Kp * Np + 255) / 256, 256, 0, stream>>>(dst, src, K, Nc, Kp, Np, tr);
  };
  conv(WiTmap, Wi_map, 64, 192, 64, 192, 1);
  conv(WhTmap, Wh_map, 64, 192, 64, 192, 1);
  conv(WiTmem, Wi_mem, 64, 192, 64, 192, 1);
  conv(WhTmem, Wh_mem, 64, 192, 64, 192, 1);
  for (int i = 0; i < 12; ++i)
    conv(wmP + (size_t)i * 96 * 80, Wm_plain + (size_t)i * 70 * 70, 70, 70, 96, 80, 0);
  for (int i = 0; i < 6; ++i)
    conv(wuP + (size_t)i * 160 * 80, Wu_plain + (size_t)i * 140 * 70, 140, 70, 160, 80, 0);
  for (int i = 0; i < 4; ++i)
    conv(wmA + (size_t)i * 160 * 144, Wm_ar + (size_t)i * 140 * 140, 140, 140, 160, 144, 0);
  for (int i = 0; i < 2; ++i)
    conv(wuA + (size_t)i * 288 * 80, Wu_ar + (size_t)i * 280 * 70, 280, 70, 288, 80, 0);
  k_convh<<<640, 256, 0, stream>>>(embedh, embed, V * 64);

  k_gru<<<(N + 63) / 64, 128, 0, stream>>>(tokens, embedh, WiTmap, WhTmap, bi_map, bh_map, hmap, N);
  k_gru<<<(N + 63) / 64, 128, 0, stream>>>(tokens, embedh, WiTmem, WhTmem, bi_mem, bh_mem, hmem, N);
  k_combine<<<1024, 256, 0, stream>>>(hmap, hmem, type_idx, bufH, N, NSTR);

  dim3 gmsg((E + 15) / 16, T);
  int  gupd = (N + 15) / 16;
  for (int blk = 0; blk < 2; ++blk) {
    hipMemcpyAsync(bufS, bufH, (size_t)N * NSTR * 2, hipMemcpyDeviceToDevice, stream);
    for (int j = 0; j < 3; ++j) {
      int li = blk * 3 + j;
      k_fillu<<<1024, 256, 0, stream>>>(aggK, N * AGGW_P, NEGKEY);
      k_msg<96, 80><<<gmsg, 128, 0, stream>>>(
          bufH, NSTR, wmP + (size_t)li * 2 * 96 * 80, bm_plain + (size_t)li * 2 * H, H,
          edge_src, edge_dst, E, H, aggK, AGGW_P, 0);
      k_build_u<<<1024, 256, 0, stream>>>(bufH, NSTR, H, aggK, AGGW_P, H, bufT, NSTR, N);
      k_upd<160, 80><<<gupd, 128, 0, stream>>>(
          bufT, NSTR, wuP + (size_t)li * 160 * 80, bu_plain + (size_t)li * H, H,
          bufH, NSTR, N, 0);
    }
    // concat-residual -> AR layer (gelu messages, tanh update)
    k_concat2<<<1024, 256, 0, stream>>>(bufS, NSTR, H, bufH, NSTR, H, bufT, NSTR, N);
    k_fillu<<<1024, 256, 0, stream>>>(aggK, N * AGGW_A, NEGKEY);
    k_msg<160, 144><<<gmsg, 128, 0, stream>>>(
        bufT, NSTR, wmA + (size_t)blk * 2 * 160 * 144, bm_ar + (size_t)blk * 2 * 140, 140,
        edge_src, edge_dst, E, 140, aggK, AGGW_A, 1);
    k_build_u<<<1024, 256, 0, stream>>>(bufT, NSTR, 140, aggK, AGGW_A, 140, ubuf, USTR, N);
    k_upd<288, 80><<<gupd, 128, 0, stream>>>(
        ubuf, USTR, wuA + (size_t)blk * 288 * 80, bu_ar + (size_t)blk * H, H,
        bufH, NSTR, N, 1);
  }
  k_head<<<1, 256, 0, stream>>>(bufH, NSTR, map_idx, W1, b1, W2, b2, out);
}
